// PROTACSplitter_90211493085314
// MI455X (gfx1250) — matrix operand order repaired and checked by reading, not compiled, not run on hardware
//
#include <hip/hip_runtime.h>

#define NNODES 200000
#define NEDGES 400000
#define NEP    100000
#define FIN    128
#define HDIM   256
#define NEGS   0.01f

typedef __attribute__((ext_vector_type(16))) __bf16       v16bf;
typedef __attribute__((ext_vector_type(8)))  float        v8f;
typedef __attribute__((ext_vector_type(4)))  float        f4;
typedef __attribute__((ext_vector_type(2)))  unsigned int u2;

__device__ __forceinline__ unsigned short f2bf(float f) {
  unsigned int u = __float_as_uint(f);
  unsigned int r = u + 0x7fffu + ((u >> 16) & 1u);   // round-to-nearest-even
  return (unsigned short)(r >> 16);
}
__device__ __forceinline__ float bf2f(unsigned short s) {
  return __uint_as_float(((unsigned int)s) << 16);
}

// ---------- utility kernels ----------
__global__ void k_zero_f32(float* __restrict__ p, int n4) {
  int i = blockIdx.x * blockDim.x + threadIdx.x;
  if (i < n4) { f4 z = {0.f, 0.f, 0.f, 0.f}; *(f4*)(p + 4 * (size_t)i) = z; }
}

__global__ void k_cvt_bf16(const float* __restrict__ in, unsigned short* __restrict__ out, int n4) {
  int i = blockIdx.x * blockDim.x + threadIdx.x;
  if (i >= n4) return;
  f4 v = *(const f4*)(in + 4 * (size_t)i);
  u2 pk;
  pk.x = (unsigned int)f2bf(v.x) | ((unsigned int)f2bf(v.y) << 16);
  pk.y = (unsigned int)f2bf(v.z) | ((unsigned int)f2bf(v.w) << 16);
  *(u2*)(out + 4 * (size_t)i) = pk;
}

// sum top/bottom halves of a (2H, H) weight:  o[k*H+n] = w[k*H+n] + w[(k+H)*H+n]
__global__ void k_addhalf(const float* __restrict__ w, float* __restrict__ o, int n) {
  int i = blockIdx.x * blockDim.x + threadIdx.x;
  if (i < n) o[i] = w[i] + w[i + n];
}

// pack W (K x Nout, row-major f32) into bf16 WMMA B-fragment order:
// frag (kt,nt) base = (kt*NT+nt)*512 ; lane l holds k = kt*32 + 16*(l>>4) + e , n = nt*16 + (l&15)
__global__ void k_pack_w(const float* __restrict__ W, unsigned short* __restrict__ out,
                         int K, int Nout) {
  int tid = blockIdx.x * blockDim.x + threadIdx.x;
  if (tid >= K * Nout) return;
  int e = tid & 15;
  int l = (tid >> 4) & 31;
  int frag = tid >> 9;
  int NT = Nout >> 4;
  int kt = frag / NT;
  int nt = frag - kt * NT;
  int k = kt * 32 + ((l >> 4) << 4) + e;
  int n = nt * 16 + (l & 15);
  out[tid] = f2bf(W[(size_t)k * Nout + n]);
}

// segment-sum: agg[dst[e]] += z[src[e]]  (bf16 in, f32 atomic accumulate), 4 feats/thread
__global__ void k_scatter(const unsigned short* __restrict__ z, int F,
                          const int* __restrict__ src, const int* __restrict__ dst,
                          float* __restrict__ agg, int total) {
  int tid = blockIdx.x * blockDim.x + threadIdx.x;
  if (tid >= total) return;
  int per = F >> 2;
  int e = tid / per;
  int f = (tid - e * per) << 2;
  int s = src[e], d = dst[e];
  u2 raw = *(const u2*)(z + (size_t)s * F + f);
  float* q = agg + (size_t)d * F + f;
  atomicAdd(q + 0, __uint_as_float((raw.x & 0xffffu) << 16));
  atomicAdd(q + 1, __uint_as_float(raw.x & 0xffff0000u));
  atomicAdd(q + 2, __uint_as_float((raw.y & 0xffffu) << 16));
  atomicAdd(q + 3, __uint_as_float(raw.y & 0xffff0000u));
}

// ---------- WMMA GEMM with LDS-resident weights ----------
//   D = act( A1@W1 + A2@W2 + bscale*bias ) (+ optional skip), Nout = 256 (NT = 16 tiles).
// Packed W panel (KT*16 fragments, 1 KB each) is staged in LDS once per block
// (128/256 KB, within the 320 KB WGP budget), then each wave grid-strides over
// PAIRS of 16-row m-tiles: every ds_load'ed B fragment feeds two v_wmma.
// B fragments are one-deep double-buffered so LDS latency hides under the WMMAs.
template <int KT1, int KT2, bool PREV, bool PCUR, bool NXT>
__global__ __launch_bounds__(256) void k_gemm(
    const unsigned short* __restrict__ A1, int K1,
    const unsigned short* __restrict__ A2, int K2,
    const unsigned short* __restrict__ Wp,
    const float* __restrict__ bias, float bscale,
    const float* __restrict__ Pprev,
    float* __restrict__ Pcur,
    unsigned short* __restrict__ nextin,
    int M, int Nout) {
  constexpr int KT = KT1 + KT2;
  constexpr int NT = 16;
  __shared__ unsigned short smem[KT * NT * 512];   // KT=16 -> 256 KB, KT=8 -> 128 KB

  // cooperative global -> LDS copy of the packed weight panel (b128 load + ds_store)
  for (int i = threadIdx.x; i < KT * NT * 64; i += blockDim.x)
    *(f4*)(smem + (size_t)i * 8) = *(const f4*)(Wp + (size_t)i * 8);
  __syncthreads();

  const int lane = threadIdx.x & 31;
  const int wave = threadIdx.x >> 5;
  const int nw = blockDim.x >> 5;
  const int h = lane >> 4;                  // K half select for A, row-half for C
  const int r = lane & 15;                  // A row within tile / C column within tile
  const int pairs = M >> 5;                 // M is a multiple of 32 for all call sites

  for (int p = blockIdx.x * nw + wave; p < pairs; p += gridDim.x * nw) {
    const int mt0 = p * 2, mt1 = p * 2 + 1;

    // Hoist all A fragments for both 16-row tiles into registers.
    v16bf a0[KT], a1[KT];
#pragma unroll
    for (int kt = 0; kt < KT; ++kt) {
      const unsigned short* b0;
      const unsigned short* b1;
      int koff;
      if (kt < KT1) {
        b0 = A1 + (size_t)(mt0 * 16 + r) * K1;
        b1 = A1 + (size_t)(mt1 * 16 + r) * K1;
        koff = kt * 32;
      } else {
        b0 = A2 + (size_t)(mt0 * 16 + r) * K2;
        b1 = A2 + (size_t)(mt1 * 16 + r) * K2;
        koff = (kt - KT1) * 32;
      }
      union { v16bf v; f4 q[2]; } u0, u1;
      u0.q[0] = *(const f4*)(b0 + koff + 16 * h);
      u0.q[1] = *(const f4*)(b0 + koff + 16 * h + 8);
      u1.q[0] = *(const f4*)(b1 + koff + 16 * h);
      u1.q[1] = *(const f4*)(b1 + koff + 16 * h + 8);
      a0[kt] = u0.v;
      a1[kt] = u1.v;
    }

    for (int nt = 0; nt < NT; ++nt) {
      v8f acc0 = {0.f, 0.f, 0.f, 0.f, 0.f, 0.f, 0.f, 0.f};
      v8f acc1 = {0.f, 0.f, 0.f, 0.f, 0.f, 0.f, 0.f, 0.f};
      // one-deep double buffer for B fragments out of LDS
      v16bf bcur = *(const v16bf*)(const void*)(smem + ((size_t)nt * 512) + lane * 16);
#pragma unroll
      for (int kt = 0; kt < KT; ++kt) {
        v16bf bnxt;
        if (kt + 1 < KT)
          bnxt = *(const v16bf*)(const void*)(smem + ((size_t)((kt + 1) * NT + nt) * 512) + lane * 16);
        acc0 = __builtin_amdgcn_wmma_f32_16x16x32_bf16(false, a0[kt], false, bcur,
                                                       (short)0, acc0, false, false);
        acc1 = __builtin_amdgcn_wmma_f32_16x16x32_bf16(false, a1[kt], false, bcur,
                                                       (short)0, acc1, false, false);
        bcur = bnxt;
      }
      const int c = nt * 16 + r;                          // output column
      const float bv = bscale * bias[c];
#pragma unroll
      for (int v = 0; v < 8; ++v) {                       // C/D: row = v + 8h, col = lane&15
        {
          int rr = mt0 * 16 + v + 8 * h;
          float val = acc0[v] + bv;
          float pre = val >= 0.f ? val : NEGS * val;      // LeakyReLU
          size_t o = (size_t)rr * Nout + c;
          if constexpr (PCUR) Pcur[o] = pre;
          float sm = pre;
          if constexpr (PREV) sm += Pprev[o];             // residual skip
          if constexpr (NXT) nextin[o] = f2bf(sm);
        }
        {
          int rr = mt1 * 16 + v + 8 * h;
          float val = acc1[v] + bv;
          float pre = val >= 0.f ? val : NEGS * val;
          size_t o = (size_t)rr * Nout + c;
          if constexpr (PCUR) Pcur[o] = pre;
          float sm = pre;
          if constexpr (PREV) sm += Pprev[o];
          if constexpr (NXT) nextin[o] = f2bf(sm);
        }
      }
    }
  }
}

// s = z[a] + z[b] with z = Pa + Pb (exact f32 skip sum), output bf16 (EP x 256)
__global__ void k_sgather(const float* __restrict__ Pa, const float* __restrict__ Pb,
                          const int* __restrict__ ep, unsigned short* __restrict__ sout,
                          int total) {
  int i = blockIdx.x * blockDim.x + threadIdx.x;
  if (i >= total) return;
  int row = i >> 8;
  int c = i & 255;
  int a = ep[2 * row], b = ep[2 * row + 1];
  float za = Pa[(size_t)a * HDIM + c] + Pb[(size_t)a * HDIM + c];
  float zb = Pa[(size_t)b * HDIM + c] + Pb[(size_t)b * HDIM + c];
  sout[i] = f2bf(za + zb);
}

// y[r,j] = h[r,:] . w1[:,j] + b1[j]
__global__ void k_headout(const unsigned short* __restrict__ hbuf, const float* __restrict__ w1,
                          const float* __restrict__ b1, int outc, float* __restrict__ out,
                          int total) {
  int tid = blockIdx.x * blockDim.x + threadIdx.x;
  if (tid >= total) return;
  int rr = tid / outc;
  int j = tid - rr * outc;
  const unsigned short* hp = hbuf + (size_t)rr * HDIM;
  float s = 0.f;
  for (int k = 0; k < HDIM; ++k) s += bf2f(hp[k]) * w1[k * outc + j];
  out[tid] = s + b1[j];
}

static inline int cdiv(long long a, long long b) { return (int)((a + b - 1) / b); }
static inline int imin(int a, int b) { return a < b ? a : b; }

extern "C" void kernel_launch(void* const* d_in, const int* in_sizes, int n_in,
                              void* d_out, int out_size, void* d_ws, size_t ws_size,
                              hipStream_t stream) {
  const float* x = (const float*)d_in[0];
  const int* ei = (const int*)d_in[1];
  const int* ep = (const int*)d_in[2];
  const float* wrel[4]  = {(const float*)d_in[3], (const float*)d_in[6], (const float*)d_in[9],  (const float*)d_in[12]};
  const float* brel[4]  = {(const float*)d_in[4], (const float*)d_in[7], (const float*)d_in[10], (const float*)d_in[13]};
  const float* wroot[4] = {(const float*)d_in[5], (const float*)d_in[8], (const float*)d_in[11], (const float*)d_in[14]};
  const float* wb0 = (const float*)d_in[15]; const float* bb0 = (const float*)d_in[16];
  const float* wb1 = (const float*)d_in[17]; const float* bb1 = (const float*)d_in[18];
  const float* wo0 = (const float*)d_in[19]; const float* bo0 = (const float*)d_in[20];
  const float* wo1 = (const float*)d_in[21]; const float* bo1 = (const float*)d_in[22];
  float* out = (float*)d_out;

  // ---- workspace carve-up ----
  char* ws = (char*)d_ws;
  const size_t NP = (size_t)NNODES * HDIM * sizeof(float);          // 204.8 MB
  const size_t NB = (size_t)NNODES * HDIM * sizeof(unsigned short); // 102.4 MB
  float* P0  = (float*)(ws);
  float* P1  = (float*)(ws + NP);
  float* AGG = (float*)(ws + 2 * NP);
  unsigned short* INA  = (unsigned short*)(ws + 3 * NP);
  unsigned short* INB  = (unsigned short*)(ws + 3 * NP + NB);
  unsigned short* AGGB = (unsigned short*)(ws + 3 * NP + 2 * NB);
  unsigned short* SB   = (unsigned short*)(ws + 3 * NP + 3 * NB);
  unsigned short* HB   = SB + (size_t)NEP * HDIM;
  char* wsm = (char*)(HB + (size_t)NEP * HDIM);
  unsigned short* WP0 = (unsigned short*)wsm; wsm += (size_t)8  * 16 * 512 * 2;
  unsigned short* WP1 = (unsigned short*)wsm; wsm += (size_t)16 * 16 * 512 * 2;
  unsigned short* WP2 = (unsigned short*)wsm; wsm += (size_t)16 * 16 * 512 * 2;
  unsigned short* WP3 = (unsigned short*)wsm; wsm += (size_t)16 * 16 * 512 * 2;
  float* WSUMF = (float*)wsm;                 wsm += (size_t)HDIM * HDIM * 4;
  unsigned short* WPHB = (unsigned short*)wsm; wsm += (size_t)8 * 16 * 512 * 2;
  unsigned short* WPHO = (unsigned short*)wsm;

  const int* src = ei;
  const int* dst = ei + NEDGES;

  // ---- pack weights into WMMA fragment order (stacked: w_rel frags then w_root frags) ----
  k_pack_w<<<cdiv(FIN * HDIM, 256), 256, 0, stream>>>(wrel[0], WP0, FIN, HDIM);
  k_pack_w<<<cdiv(FIN * HDIM, 256), 256, 0, stream>>>(wroot[0], WP0 + 4 * 16 * 512, FIN, HDIM);
  unsigned short* WPs[4] = {WP0, WP1, WP2, WP3};
  for (int L = 1; L < 4; ++L) {
    k_pack_w<<<cdiv(HDIM * HDIM, 256), 256, 0, stream>>>(wrel[L], WPs[L], HDIM, HDIM);
    k_pack_w<<<cdiv(HDIM * HDIM, 256), 256, 0, stream>>>(wroot[L], WPs[L] + 8 * 16 * 512, HDIM, HDIM);
  }
  // heads: fold ze0@w0 + ze1@w0 = (za+zb) @ (w0_top + w0_bot)
  k_addhalf<<<cdiv(HDIM * HDIM, 256), 256, 0, stream>>>(wb0, WSUMF, HDIM * HDIM);
  k_pack_w<<<cdiv(HDIM * HDIM, 256), 256, 0, stream>>>(WSUMF, WPHB, HDIM, HDIM);
  k_addhalf<<<cdiv(HDIM * HDIM, 256), 256, 0, stream>>>(wo0, WSUMF, HDIM * HDIM);
  k_pack_w<<<cdiv(HDIM * HDIM, 256), 256, 0, stream>>>(WSUMF, WPHO, HDIM, HDIM);

  // ---- x -> bf16 (layer-0 input, row stride 128) ----
  k_cvt_bf16<<<cdiv((size_t)NNODES * FIN / 4, 256), 256, 0, stream>>>(x, INA, NNODES * FIN / 4);

  unsigned short* inc = INA;
  unsigned short* inn = INB;
  for (int L = 0; L < 4; ++L) {
    const int K = (L == 0) ? FIN : HDIM;
    const int nK = NNODES * K;
    k_zero_f32<<<cdiv(nK / 4, 256), 256, 0, stream>>>(AGG, nK / 4);
    const int tot = NEDGES * (K >> 2);
    k_scatter<<<cdiv(tot, 256), 256, 0, stream>>>(inc, K, src, dst, AGG, tot);
    k_cvt_bf16<<<cdiv(nK / 4, 256), 256, 0, stream>>>(AGG, AGGB, nK / 4);

    float* Pc = (L & 1) ? P1 : P0;                       // L0->P0, L1->P1, L2->P0, L3->P1
    float* Pp = (L == 0) ? nullptr : ((L & 1) ? P0 : P1);
    const int blocks = imin(640, cdiv(NNODES / 32, 8));  // grid-stride over m-tile pairs
    if (L == 0)
      k_gemm<4, 4, false, true, true><<<blocks, 256, 0, stream>>>(
          AGGB, K, inc, K, WPs[0], brel[0], 1.f, Pp, Pc, inn, NNODES, HDIM);
    else
      k_gemm<8, 8, true, true, true><<<blocks, 256, 0, stream>>>(
          AGGB, K, inc, K, WPs[L], brel[L], 1.f, Pp, Pc, inn, NNODES, HDIM);
    unsigned short* t = inc; inc = inn; inn = t;
  }

  // ---- heads: s = z[a]+z[b], z = P1 + P0 (final skip sum) ----
  k_sgather<<<cdiv((size_t)NEP * HDIM, 256), 256, 0, stream>>>(P1, P0, ep, SB, NEP * HDIM);
  const int hblocks = imin(640, cdiv(NEP / 32, 8));
  // location head (wb*, outc=1) -> d_out[3*EP ..)
  k_gemm<8, 0, false, false, true><<<hblocks, 256, 0, stream>>>(
      SB, HDIM, (const unsigned short*)nullptr, 0, WPHB, bb0, 2.f, nullptr, nullptr, HB, NEP, HDIM);
  k_headout<<<cdiv(NEP, 256), 256, 0, stream>>>(HB, wb1, bb1, 1, out + (size_t)NEP * 3, NEP);
  // type head (wo*, outc=3) -> d_out[0 ..)
  k_gemm<8, 0, false, false, true><<<hblocks, 256, 0, stream>>>(
      SB, HDIM, (const unsigned short*)nullptr, 0, WPHO, bo0, 2.f, nullptr, nullptr, HB, NEP, HDIM);
  k_headout<<<cdiv(NEP * 3, 256), 256, 0, stream>>>(HB, wo1, bo1, 3, out, NEP * 3);
}
